// DotAttention_68736656605628
// MI455X (gfx1250) — matrix operand order repaired and checked by reading, not compiled, not run on hardware
//
#include <hip/hip_runtime.h>

typedef __attribute__((ext_vector_type(16))) __bf16 v16bf;
typedef __attribute__((ext_vector_type(8)))  float  v8f;

#define B_     16
#define LQ_    2048
#define LK_    2048
#define D_     64
#define QTILE  64      // q rows per workgroup
#define KSTEP  32      // keys per iteration (matches bf16 WMMA K-dim)
#define NWAVES 4
#define L2E    1.4426950408889634f
#define NEGMAX (-3.0e38f)

template<int M>
__device__ __forceinline__ float swz_xor(float x) {
  // ds_swizzle group-of-32: and=0x1f, or=0, xor=M  -> src lane = lane ^ M
  return __builtin_bit_cast(float,
      __builtin_amdgcn_ds_swizzle(__builtin_bit_cast(int, x), 0x1f | (M << 10)));
}

__device__ __forceinline__ v8f wmma_bf16(v16bf a, v16bf b, v8f c) {
  return __builtin_amdgcn_wmma_f32_16x16x32_bf16(false, a, false, b,
                                                 (short)0, c, false, false);
}

// Low 32 bits of a generic pointer to __shared__ == LDS byte offset
__device__ __forceinline__ unsigned lds_addr32(const void* p) {
  return (unsigned)(unsigned long long)p;
}

// gfx1250 async memory->LDS copy: each lane moves 16B, tracked by ASYNCcnt.
__device__ __forceinline__ void async_copy_b128(unsigned lds, const float* g) {
  asm volatile("global_load_async_to_lds_b128 %0, %1, off"
               :: "v"(lds), "v"(g) : "memory");
}

template<int N>
__device__ __forceinline__ void wait_asynccnt() {
  asm volatile("s_wait_asynccnt %0" :: "n"(N) : "memory");
}

// Issue async staging of one 32x64 f32 K block and V block (8 x b128 / thread).
__device__ __forceinline__ void issue_block(const float* Kb, const float* Vb,
                                            float* rK, float* rV, int tid) {
  #pragma unroll
  for (int it = 0; it < 4; ++it) {
    const int idx = (tid + 128 * it) << 2;     // float index of 16B chunk
    async_copy_b128(lds_addr32(rK + idx), Kb + idx);
    async_copy_b128(lds_addr32(rV + idx), Vb + idx);
  }
}

__global__ __launch_bounds__(128, 1)
void DotAttention_68736656605628_kernel(const float* __restrict__ q,
                                        const float* __restrict__ k,
                                        const float* __restrict__ v,
                                        const float* __restrict__ vmask,
                                        float* __restrict__ out)
{
  __shared__ __align__(16) float  sRawK[2][KSTEP * D_];  // async-staged f32
  __shared__ __align__(16) float  sRawV[2][KSTEP * D_];
  __shared__ __align__(32) __bf16 sKhi[KSTEP * D_];      // K row-major [key][d]
  __shared__ __align__(32) __bf16 sKlo[KSTEP * D_];
  __shared__ __align__(32) __bf16 sVhi[D_ * KSTEP];      // V transposed [d][key]
  __shared__ __align__(32) __bf16 sVlo[D_ * KSTEP];
  __shared__ float sMask[KSTEP];
  __shared__ __align__(32) float sP[NWAVES][16 * KSTEP]; // per-wave P re-layout

  const int tid  = threadIdx.x;
  const int lane = tid & 31;
  const int wv   = tid >> 5;
  const int b    = blockIdx.y;
  const int q0   = blockIdx.x * QTILE;
  const int nrow = lane & 15;          // A-frag row / B-frag column / C-frag N
  const int half = lane >> 4;
  const int kb8  = half ? 8 : 0;       // unused; A- and B-frags both use kb16
  const int kb16 = half ? 16 : 0;      // B-frag K base

  const float* Kbase = k + (size_t)b * LK_ * D_;
  const float* Vbase = v + (size_t)b * LK_ * D_;

  // ---- Q A-fragments (16x32 per D-chunk), 2-way bf16 split, loaded once ----
  const float* Qr = q + ((size_t)b * LQ_ + q0 + wv * 16 + nrow) * D_;
  v16bf a_hi[2], a_lo[2];
  #pragma unroll
  for (int c = 0; c < 2; ++c) {
    const int base = 32 * c + kb16;
    #pragma unroll
    for (int i = 0; i < 8; ++i) {
      float  f0 = Qr[base + i];
      __bf16 h0 = (__bf16)f0;
      a_hi[c][i] = h0;            a_lo[c][i]     = (__bf16)(f0 - (float)h0);
      float  f1 = Qr[base + 8 + i];
      __bf16 h1 = (__bf16)f1;
      a_hi[c][8 + i] = h1;        a_lo[c][8 + i] = (__bf16)(f1 - (float)h1);
    }
  }

  v8f o[4];
  #pragma unroll
  for (int dt = 0; dt < 4; ++dt) o[dt] = (v8f)(0.0f);
  float mrow[8], srow[8];
  #pragma unroll
  for (int j = 0; j < 8; ++j) { mrow[j] = NEGMAX; srow[j] = 0.0f; }

  // Prologue: async-stage block 0 into raw buffer 0.
  issue_block(Kbase, Vbase, sRawK[0], sRawV[0], tid);

  for (int kt = 0; kt < LK_; kt += KSTEP) {
    const int cur = (kt / KSTEP) & 1;

    // Early mask load for this block (overlaps the async wait).
    float mval = 0.0f;
    if (tid < KSTEP) mval = vmask[(size_t)b * LK_ + kt + tid];

    // Prefetch next block into the other raw buffer, then wait for current.
    if (kt + KSTEP < LK_) {
      issue_block(Kbase + (size_t)(kt + KSTEP) * D_,
                  Vbase + (size_t)(kt + KSTEP) * D_,
                  sRawK[cur ^ 1], sRawV[cur ^ 1], tid);
      wait_asynccnt<8>();    // in-order: current block's 8 copies are done
    } else {
      wait_asynccnt<0>();
    }
    __syncthreads();         // raw block visible to all waves; prev iter done

    // ---- convert raw f32 -> bf16 hi/lo (V transposed) ----
    const float* rK = sRawK[cur];
    const float* rV = sRawV[cur];
    #pragma unroll
    for (int it = 0; it < 4; ++it) {
      int idx = tid + 128 * it;        // 0..511
      int row = idx >> 4;              // key 0..31
      int c4  = (idx & 15) << 2;       // dim 0..60
      float4 k4 = *(const float4*)(rK + row * D_ + c4);
      float4 v4 = *(const float4*)(rV + row * D_ + c4);
      const float kf[4] = {k4.x, k4.y, k4.z, k4.w};
      const float vf[4] = {v4.x, v4.y, v4.z, v4.w};
      #pragma unroll
      for (int jj = 0; jj < 4; ++jj) {
        __bf16 kh = (__bf16)kf[jj];
        sKhi[row * D_ + c4 + jj] = kh;
        sKlo[row * D_ + c4 + jj] = (__bf16)(kf[jj] - (float)kh);
        __bf16 vh = (__bf16)vf[jj];
        sVhi[(c4 + jj) * KSTEP + row] = vh;
        sVlo[(c4 + jj) * KSTEP + row] = (__bf16)(vf[jj] - (float)vh);
      }
    }
    if (tid < KSTEP) sMask[tid] = mval;
    __syncthreads();

    // ---- S = Q * K^T : 2 key tiles x 2 D-chunks x 3 bf16 terms = 12 WMMA ----
    v8f sfrag[2];
    #pragma unroll
    for (int t = 0; t < 2; ++t) {
      v8f c = (v8f)(0.0f);
      #pragma unroll
      for (int cc = 0; cc < 2; ++cc) {
        const v16bf bhi = *(const v16bf*)&sKhi[(t * 16 + nrow) * D_ + 32 * cc + kb16];
        const v16bf blo = *(const v16bf*)&sKlo[(t * 16 + nrow) * D_ + 32 * cc + kb16];
        c = wmma_bf16(a_hi[cc], bhi, c);
        c = wmma_bf16(a_hi[cc], blo, c);
        c = wmma_bf16(a_lo[cc], bhi, c);
      }
      sfrag[t] = c;
    }

    // ---- masked online softmax, reductions batched for ILP ----
    const float mk0 = sMask[nrow];
    const float mk1 = sMask[16 + nrow];
    float mx[8], ls[8];
    #pragma unroll
    for (int j = 0; j < 8; ++j) {
      float e0 = (mk0 != 0.0f) ? sfrag[0][j] : NEGMAX;
      float e1 = (mk1 != 0.0f) ? sfrag[1][j] : NEGMAX;
      mx[j] = fmaxf(e0, e1);
    }
    #pragma unroll
    for (int j = 0; j < 8; ++j) mx[j] = fmaxf(mx[j], swz_xor<1>(mx[j]));
    #pragma unroll
    for (int j = 0; j < 8; ++j) mx[j] = fmaxf(mx[j], swz_xor<2>(mx[j]));
    #pragma unroll
    for (int j = 0; j < 8; ++j) mx[j] = fmaxf(mx[j], swz_xor<4>(mx[j]));
    #pragma unroll
    for (int j = 0; j < 8; ++j) mx[j] = fmaxf(mx[j], swz_xor<8>(mx[j]));
    #pragma unroll
    for (int j = 0; j < 8; ++j) {
      const float mnew = fmaxf(mrow[j], mx[j]);
      const float al = __builtin_exp2f((mrow[j] - mnew) * L2E);
      const float p0 = (mk0 != 0.0f)
                         ? __builtin_exp2f((sfrag[0][j] - mnew) * L2E) : 0.0f;
      const float p1 = (mk1 != 0.0f)
                         ? __builtin_exp2f((sfrag[1][j] - mnew) * L2E) : 0.0f;
      ls[j] = p0 + p1;
      srow[j] *= al;
      mrow[j] = mnew;
      #pragma unroll
      for (int dt = 0; dt < 4; ++dt) o[dt][j] *= al;   // rescale accumulators
      const int r = j + (half << 3);
      sP[wv][r * KSTEP + nrow]      = p0;              // C-layout -> LDS
      sP[wv][r * KSTEP + 16 + nrow] = p1;
    }
    #pragma unroll
    for (int j = 0; j < 8; ++j) ls[j] += swz_xor<1>(ls[j]);
    #pragma unroll
    for (int j = 0; j < 8; ++j) ls[j] += swz_xor<2>(ls[j]);
    #pragma unroll
    for (int j = 0; j < 8; ++j) ls[j] += swz_xor<4>(ls[j]);
    #pragma unroll
    for (int j = 0; j < 8; ++j) ls[j] += swz_xor<8>(ls[j]);
    #pragma unroll
    for (int j = 0; j < 8; ++j) srow[j] += ls[j];
    __syncthreads();

    // ---- O += P * V : read P back in A-frag layout, split hi/lo ----
    v16bf phi, plo;
    const float* Pr = &sP[wv][nrow * KSTEP];
    #pragma unroll
    for (int i = 0; i < 8; ++i) {
      float  f0 = Pr[kb16 + i];
      __bf16 h0 = (__bf16)f0;
      phi[i] = h0;          plo[i]     = (__bf16)(f0 - (float)h0);
      float  f1 = Pr[kb16 + 8 + i];
      __bf16 h1 = (__bf16)f1;
      phi[8 + i] = h1;      plo[8 + i] = (__bf16)(f1 - (float)h1);
    }
    #pragma unroll
    for (int dt = 0; dt < 4; ++dt) {
      const v16bf vhi = *(const v16bf*)&sVhi[(dt * 16 + nrow) * KSTEP + kb16];
      const v16bf vlo = *(const v16bf*)&sVlo[(dt * 16 + nrow) * KSTEP + kb16];
      o[dt] = wmma_bf16(phi, vhi, o[dt]);
      o[dt] = wmma_bf16(phi, vlo, o[dt]);
      o[dt] = wmma_bf16(plo, vhi, o[dt]);
    }
  }

  // ---- normalize (matches reference: sum + 1e-13) and write out ----
  #pragma unroll
  for (int j = 0; j < 8; ++j) {
    const float inv = 1.0f / (srow[j] + 1e-13f);
    const int r = j + (half << 3);
    float* Or = out + ((size_t)b * LQ_ + q0 + wv * 16 + r) * D_ + nrow;
    #pragma unroll
    for (int dt = 0; dt < 4; ++dt) Or[dt * 16] = o[dt][j] * inv;
  }
}

extern "C" void kernel_launch(void* const* d_in, const int* in_sizes, int n_in,
                              void* d_out, int out_size, void* d_ws, size_t ws_size,
                              hipStream_t stream) {
  const float* q  = (const float*)d_in[0];
  const float* k  = (const float*)d_in[1];
  const float* v  = (const float*)d_in[2];
  const float* vm = (const float*)d_in[3];
  float* o = (float*)d_out;
  dim3 grid(LQ_ / QTILE, B_);
  dim3 block(128);
  hipLaunchKernelGGL(DotAttention_68736656605628_kernel, grid, block, 0, stream,
                     q, k, v, vm, o);
}
